// RN_43370579755072
// MI455X (gfx1250) — matrix operand order repaired and checked by reading, not compiled, not run on hardware
//
#include <hip/hip_runtime.h>
#include <hip/hip_bf16.h>

// ---------------------------------------------------------------------------
// Types for CDNA5 WMMA (wave32)
// ---------------------------------------------------------------------------
typedef __bf16        v16bf __attribute__((ext_vector_type(16)));
typedef float         v8f   __attribute__((ext_vector_type(8)));
typedef unsigned int  u32x4 __attribute__((ext_vector_type(4)));

union Frag32B { v16bf v; u32x4 q[2]; };

__device__ __forceinline__ unsigned short f2bf(float f) {
    unsigned int u = __float_as_uint(f);
    unsigned int r = u + 0x7FFFu + ((u >> 16) & 1u);   // round-to-nearest-even
    return (unsigned short)(r >> 16);
}
__device__ __forceinline__ float bf2f(unsigned short h) {
    return __uint_as_float(((unsigned int)h) << 16);
}

// ---------------------------------------------------------------------------
// Utility kernels
// ---------------------------------------------------------------------------
__global__ void rn_zero_f32(float* p, int n) {
    int i = blockIdx.x * 256 + threadIdx.x;
    if (i < n) p[i] = 0.0f;
}

__global__ void rn_f32_to_bf16(const float* __restrict__ src,
                               unsigned short* __restrict__ dst, int n) {
    int i = blockIdx.x * 256 + threadIdx.x;
    if (i < n) dst[i] = f2bf(src[i]);
}

// Convert f32 weight [N][Kin] -> bf16 [N][Kpad] (zero padded along K)
__global__ void rn_w_to_bf16(const float* __restrict__ Wf,
                             unsigned short* __restrict__ Wb,
                             int N, int Kin, int Kpad) {
    int i = blockIdx.x * 256 + threadIdx.x;
    if (i >= N * Kpad) return;
    int n = i / Kpad, k = i % Kpad;
    Wb[i] = (k < Kin) ? f2bf(Wf[n * Kin + k]) : (unsigned short)0;
}

// ---------------------------------------------------------------------------
// Direct conv 3x3 stride2 pad1 + bias + ReLU, with per-channel batch-stat
// accumulation (sum, sumsq) via block LDS reduction + 2 atomics per block.
// grid: (ceil(HW/256), B*Cout), block: 256
// ---------------------------------------------------------------------------
__global__ __launch_bounds__(256) void rn_conv_relu_stats(
    const float* __restrict__ X, const float* __restrict__ Wc,
    const float* __restrict__ bias, float* __restrict__ Y,
    float* __restrict__ stats,
    int Cin, int Hin, int Win, int Cout, int Hout, int Wout) {
    const int hw  = Hout * Wout;
    const int pos = blockIdx.x * 256 + threadIdx.x;
    const int by  = blockIdx.y;                 // b*Cout + co
    const int b   = by / Cout, co = by % Cout;
    float v = 0.0f;
    if (pos < hw) {
        const int ho = pos / Wout, wo = pos % Wout;
        const int hi0 = ho * 2 - 1, wi0 = wo * 2 - 1;
        float acc = bias[co];
        for (int ci = 0; ci < Cin; ++ci) {
            const float* xp = X + (size_t)(b * Cin + ci) * Hin * Win;
            const float* wp = Wc + (size_t)(co * Cin + ci) * 9;
#pragma unroll
            for (int kh = 0; kh < 3; ++kh) {
                int h = hi0 + kh;
                if (h < 0 || h >= Hin) continue;
#pragma unroll
                for (int kw = 0; kw < 3; ++kw) {
                    int w = wi0 + kw;
                    if (w < 0 || w >= Win) continue;
                    acc += xp[h * Win + w] * wp[kh * 3 + kw];
                }
            }
        }
        v = fmaxf(acc, 0.0f);
        Y[(size_t)by * hw + pos] = v;
    }
    __shared__ float s1[256];
    __shared__ float s2[256];
    s1[threadIdx.x] = v;
    s2[threadIdx.x] = v * v;
    __syncthreads();
    for (int s = 128; s > 0; s >>= 1) {
        if (threadIdx.x < s) {
            s1[threadIdx.x] += s1[threadIdx.x + s];
            s2[threadIdx.x] += s2[threadIdx.x + s];
        }
        __syncthreads();
    }
    if (threadIdx.x == 0) {
        atomicAdd(&stats[co * 2 + 0], s1[0]);
        atomicAdd(&stats[co * 2 + 1], s2[0]);
    }
}

// ---------------------------------------------------------------------------
// BatchNorm apply (training-mode batch stats), in place.
// ---------------------------------------------------------------------------
__global__ __launch_bounds__(256) void rn_bn_apply(
    float* __restrict__ Y, const float* __restrict__ stats,
    const float* __restrict__ gamma, const float* __restrict__ beta,
    int C, int HW, long long total, float invCnt) {
    long long i = (long long)blockIdx.x * 256 + threadIdx.x;
    if (i >= total) return;
    int c = (int)((i / (long long)HW) % C);
    float mu  = stats[c * 2 + 0] * invCnt;
    float var = stats[c * 2 + 1] * invCnt - mu * mu;
    float inv = rsqrtf(var + 1e-5f);
    Y[i] = gamma[c] * (Y[i] - mu) * inv + beta[c];
}

// ---------------------------------------------------------------------------
// Build pair matrix: P[b*625 + j*25 + i][64] bf16 = [x_i(26) | x_j(26) | q(11) | 0]
// X4 = normalized conv4 output, layout [B][24][25].
// ---------------------------------------------------------------------------
__global__ __launch_bounds__(256) void rn_build_pairs(
    const float* __restrict__ X4, const float* __restrict__ qst,
    unsigned short* __restrict__ P) {
    int r = blockIdx.x * 256 + threadIdx.x;
    if (r >= 1024 * 625) return;
    const int i = r % 25;
    const int j = (r / 25) % 25;
    const int b = r / 625;
    unsigned short* row = P + (size_t)r * 64;
    const float* xb = X4 + (size_t)b * 24 * 25;
#pragma unroll
    for (int c = 0; c < 24; ++c) row[c]      = f2bf(xb[c * 25 + i]);
    row[24] = f2bf((i * 0.2f - 2.0f) * 0.5f);          // (i/5.0 - 2)/2, true div
    row[25] = f2bf(((float)(i % 5) - 2.0f) * 0.5f);
#pragma unroll
    for (int c = 0; c < 24; ++c) row[26 + c] = f2bf(xb[c * 25 + j]);
    row[50] = f2bf((j * 0.2f - 2.0f) * 0.5f);
    row[51] = f2bf(((float)(j % 5) - 2.0f) * 0.5f);
    const float* q = qst + b * 11;
#pragma unroll
    for (int c = 0; c < 11; ++c) row[52 + c] = f2bf(q[c]);
    row[63] = 0;
}

// ---------------------------------------------------------------------------
// bf16 WMMA GEMM:  out = relu?( A[M][K] * W[256][K]^T + bias )
// K compile-time (64/256): all fragment addresses fold into 24-bit instruction
// offsets (one base pointer each for A and W). Wave tile: 16 rows x 256 cols;
// block: 8 waves = 128 rows => each A element read from global exactly once.
// Inner loop flattened over t = kstep*16 + nfrag, fully unrolled, with a
// depth-4 rotating B-fragment buffer (refill issued 4 WMMAs ahead) and a
// ping-pong A fragment (issued 16 WMMAs ahead). __builtin_amdgcn_sched_barrier
// after each iteration pins the issue order so the scheduler cannot sink the
// refill loads back to their consumers (which would re-coalesce the buffers
// and serialize every WMMA on loadcnt==0, as seen in earlier rounds).
// If sumOut != null: ReLU + f32 atomic accumulate into sumOut[(m/625)*256+n]
// (fused g4 + pair-sum). M multiple of 128. Grid: (M/128).
// ---------------------------------------------------------------------------
template <int K>
__global__ __launch_bounds__(256) void rn_gemm_bf16_wmma(
    const unsigned short* __restrict__ A,   // [M][K] bf16 bits
    const unsigned short* __restrict__ W,   // [256][K] bf16 bits
    const float* __restrict__ bias,         // [256]
    unsigned short* __restrict__ C,         // [M][256] bf16 bits (if sumOut==0)
    float* __restrict__ sumOut,             // [M/625][256] f32 accum (optional)
    int M, int relu) {
    constexpr int N      = 256;
    constexpr int KSTEPS = K / 32;
    constexpr int T      = KSTEPS * 16;     // total fragment-products per wave
    const int lane  = threadIdx.x & 31;
    const int wave  = threadIdx.x >> 5;
    const int hi    = lane >> 4;            // half-wave id
    const int ln    = lane & 15;
    const int mbase = blockIdx.x * 128 + wave * 16;
    const int row   = mbase + ln;

    v8f acc[16];
#pragma unroll
    for (int f = 0; f < 16; ++f)
#pragma unroll
        for (int r = 0; r < 8; ++r) acc[f][r] = 0.0f;

    const unsigned short* Arow = A + (size_t)row * K + hi * 8;
    const unsigned short* Wcol = W + (size_t)ln * K + hi * 8;

    Frag32B aF[2];
    Frag32B bF[4];

    // B-fragment t: kstep = t>>4, nfrag = t&15 (constant offsets after unroll)
#define RN_LOAD_B(dst, t)                                                     \
    do {                                                                      \
        (dst).q[0] = *(const u32x4*)(Wcol + ((t) & 15) * 16 * K + ((t) >> 4) * 32);     \
        (dst).q[1] = *(const u32x4*)(Wcol + ((t) & 15) * 16 * K + ((t) >> 4) * 32 + 16); \
    } while (0)
#define RN_LOAD_A(dst, ks)                                                    \
    do {                                                                      \
        (dst).q[0] = *(const u32x4*)(Arow + (ks) * 32);                       \
        (dst).q[1] = *(const u32x4*)(Arow + (ks) * 32 + 16);                  \
    } while (0)

    RN_LOAD_A(aF[0], 0);
    RN_LOAD_B(bF[0], 0);
    RN_LOAD_B(bF[1], 1);
    RN_LOAD_B(bF[2], 2);
    RN_LOAD_B(bF[3], 3);
    __builtin_amdgcn_sched_barrier(0);      // keep priming loads up front

#pragma unroll
    for (int t = 0; t < T; ++t) {
        const int ks = t >> 4;
        const int f  = t & 15;
        if (f == 0 && ks + 1 < KSTEPS) RN_LOAD_A(aF[(ks + 1) & 1], ks + 1);
        acc[f] = __builtin_amdgcn_wmma_f32_16x16x32_bf16(
            false, aF[ks & 1].v, false, bF[t & 3].v, (short)0, acc[f],
            false, false);
        if (t + 4 < T) RN_LOAD_B(bF[t & 3], t + 4);   // refill just-consumed slot
        __builtin_amdgcn_sched_barrier(0);            // pin pipeline order
    }
#undef RN_LOAD_B
#undef RN_LOAD_A

    if (sumOut) {
        // fused ReLU + per-batch (625 rows) reduction via f32 atomics
        int bidx[8];
#pragma unroll
        for (int r = 0; r < 8; ++r) bidx[r] = (mbase + hi * 8 + r) / 625;
#pragma unroll
        for (int f = 0; f < 16; ++f) {
            const int n = f * 16 + ln;
            const float bs = bias[n];
#pragma unroll
            for (int r = 0; r < 8; ++r) {
                float v = fmaxf(acc[f][r] + bs, 0.0f);
                atomicAdd(&sumOut[(size_t)bidx[r] * N + n], v);
            }
        }
    } else {
#pragma unroll
        for (int f = 0; f < 16; ++f) {
            const int n = f * 16 + ln;
            const float bs = bias[n];
#pragma unroll
            for (int r = 0; r < 8; ++r) {
                const int m = mbase + hi * 8 + r;
                float v = acc[f][r] + bs;
                if (relu) v = fmaxf(v, 0.0f);
                C[(size_t)m * N + n] = f2bf(v);
            }
        }
    }
}

// ---------------------------------------------------------------------------
// Final 256 -> 10 head, f32 output.
// ---------------------------------------------------------------------------
__global__ __launch_bounds__(256) void rn_final_fc(
    const unsigned short* __restrict__ H, const float* __restrict__ W,
    const float* __restrict__ bias, float* __restrict__ out) {
    int i = blockIdx.x * 256 + threadIdx.x;   // b*10 + n
    if (i >= 1024 * 10) return;
    const int b = i / 10, n = i % 10;
    const unsigned short* h = H + (size_t)b * 256;
    const float* w = W + n * 256;
    float s = bias[n];
    for (int k = 0; k < 256; ++k) s += bf2f(h[k]) * w[k];
    out[i] = s;
}

// ---------------------------------------------------------------------------
// Host-side orchestration
// ---------------------------------------------------------------------------
extern "C" void kernel_launch(void* const* d_in, const int* in_sizes, int n_in,
                              void* d_out, int out_size, void* d_ws, size_t ws_size,
                              hipStream_t stream) {
    (void)in_sizes; (void)n_in; (void)out_size; (void)ws_size;
    const float* img = (const float*)d_in[0];
    const float* qst = (const float*)d_in[1];
    const float* cw[4] = {(const float*)d_in[2], (const float*)d_in[4],
                          (const float*)d_in[6], (const float*)d_in[8]};
    const float* cb[4] = {(const float*)d_in[3], (const float*)d_in[5],
                          (const float*)d_in[7], (const float*)d_in[9]};
    const float* bg[4] = {(const float*)d_in[10], (const float*)d_in[12],
                          (const float*)d_in[14], (const float*)d_in[16]};
    const float* bb[4] = {(const float*)d_in[11], (const float*)d_in[13],
                          (const float*)d_in[15], (const float*)d_in[17]};
    const float* gw[4] = {(const float*)d_in[18], (const float*)d_in[20],
                          (const float*)d_in[22], (const float*)d_in[24]};
    const float* gb[4] = {(const float*)d_in[19], (const float*)d_in[21],
                          (const float*)d_in[23], (const float*)d_in[25]};
    const float* fw1 = (const float*)d_in[26]; const float* fb1 = (const float*)d_in[27];
    const float* fw2 = (const float*)d_in[28]; const float* fb2 = (const float*)d_in[29];
    const float* fw3 = (const float*)d_in[30]; const float* fb3 = (const float*)d_in[31];
    float* out = (float*)d_out;

    // ---- workspace layout (aliasing: conv1 lives in hA, conv2/conv3 in hB) ----
    char* ws = (char*)d_ws;
    size_t off = 0;
    auto take = [&](size_t bytes) { char* p = ws + off; off = (off + bytes + 255) & ~(size_t)255; return p; };
    const size_t szPairs = (size_t)640000 * 64 * 2;    // 82 MB
    const size_t szH     = (size_t)640000 * 256 * 2;   // 328 MB
    unsigned short* pairs = (unsigned short*)take(szPairs);
    char* hA = take(szH);                               // g-MLP ping (aliases conv1 out, 142MB)
    char* hB = take(szH);                               // g-MLP pong (aliases conv2/conv3 out)
    float* conv4 = (float*)take((size_t)1024 * 24 * 25 * 4);
    float* stats = (float*)take(4 * 24 * 2 * sizeof(float));
    unsigned short* wbG1 = (unsigned short*)take(256 * 64 * 2);
    unsigned short* wbG2 = (unsigned short*)take(256 * 256 * 2);
    unsigned short* wbG3 = (unsigned short*)take(256 * 256 * 2);
    unsigned short* wbG4 = (unsigned short*)take(256 * 256 * 2);
    unsigned short* wbF1 = (unsigned short*)take(256 * 256 * 2);
    unsigned short* wbF2 = (unsigned short*)take(256 * 256 * 2);
    float*          sumf = (float*)take((size_t)1024 * 256 * 4);
    unsigned short* sumb = (unsigned short*)take((size_t)1024 * 256 * 2);
    unsigned short* f1b  = (unsigned short*)take((size_t)1024 * 256 * 2);
    unsigned short* f2b  = (unsigned short*)take((size_t)1024 * 256 * 2);

    float* conv1 = (float*)hA;                               // 1024*24*38*38 f32
    float* conv2 = (float*)hB;                               // 1024*24*19*19 f32
    float* conv3 = (float*)(hB + (size_t)160 * 1024 * 1024); // 1024*24*10*10 f32

    // ---- stats/sum zero + weight conversion ----
    rn_zero_f32<<<1, 256, 0, stream>>>(stats, 192);
    rn_zero_f32<<<(1024 * 256 + 255) / 256, 256, 0, stream>>>(sumf, 1024 * 256);
    rn_w_to_bf16<<<(256 * 64 + 255) / 256, 256, 0, stream>>>(gw[0], wbG1, 256, 63, 64);
    rn_w_to_bf16<<<(256 * 256 + 255) / 256, 256, 0, stream>>>(gw[1], wbG2, 256, 256, 256);
    rn_w_to_bf16<<<(256 * 256 + 255) / 256, 256, 0, stream>>>(gw[2], wbG3, 256, 256, 256);
    rn_w_to_bf16<<<(256 * 256 + 255) / 256, 256, 0, stream>>>(gw[3], wbG4, 256, 256, 256);
    rn_w_to_bf16<<<(256 * 256 + 255) / 256, 256, 0, stream>>>(fw1, wbF1, 256, 256, 256);
    rn_w_to_bf16<<<(256 * 256 + 255) / 256, 256, 0, stream>>>(fw2, wbF2, 256, 256, 256);

    // ---- conv stack: (X, Y, Cin, Hin, Win, Hout, Wout) ----
    struct CL { const float* X; float* Y; int Cin, Hin, Win, Hout, Wout; };
    CL layers[4] = {
        { img,   conv1,  3, 75, 75, 38, 38 },
        { conv1, conv2, 24, 38, 38, 19, 19 },
        { conv2, conv3, 24, 19, 19, 10, 10 },
        { conv3, conv4, 24, 10, 10,  5,  5 },
    };
    for (int l = 0; l < 4; ++l) {
        const CL& L = layers[l];
        const int HW = L.Hout * L.Wout;
        dim3 g((HW + 255) / 256, 1024 * 24);
        rn_conv_relu_stats<<<g, 256, 0, stream>>>(
            L.X, cw[l], cb[l], L.Y, stats + l * 48,
            L.Cin, L.Hin, L.Win, 24, L.Hout, L.Wout);
        long long tot = (long long)1024 * 24 * HW;
        rn_bn_apply<<<(unsigned)((tot + 255) / 256), 256, 0, stream>>>(
            L.Y, stats + l * 48, bg[l], bb[l], 24, HW, tot,
            1.0f / (1024.0f * (float)HW));
    }

    // ---- pair construction: 640000 x 64 bf16 ----
    rn_build_pairs<<<(1024 * 625 + 255) / 256, 256, 0, stream>>>(conv4, qst, pairs);

    // ---- g-MLP (WMMA bf16, M = 640000, block covers 128 rows) ----
    const int Mg = 640000;
    rn_gemm_bf16_wmma<64><<<Mg / 128, 256, 0, stream>>>(pairs, wbG1, gb[0],
                                                        (unsigned short*)hA, nullptr, Mg, 1);
    rn_gemm_bf16_wmma<256><<<Mg / 128, 256, 0, stream>>>((unsigned short*)hA, wbG2, gb[1],
                                                         (unsigned short*)hB, nullptr, Mg, 1);
    rn_gemm_bf16_wmma<256><<<Mg / 128, 256, 0, stream>>>((unsigned short*)hB, wbG3, gb[2],
                                                         (unsigned short*)hA, nullptr, Mg, 1);
    // g4: fused ReLU + sum over the 625 pairs of each batch element
    rn_gemm_bf16_wmma<256><<<Mg / 128, 256, 0, stream>>>((unsigned short*)hA, wbG4, gb[3],
                                                         nullptr, sumf, Mg, 1);
    rn_f32_to_bf16<<<(1024 * 256 + 255) / 256, 256, 0, stream>>>(sumf, sumb, 1024 * 256);

    // ---- f-MLP (WMMA bf16, M = 1024) + head ----
    rn_gemm_bf16_wmma<256><<<1024 / 128, 256, 0, stream>>>(sumb, wbF1, fb1, f1b, nullptr, 1024, 1);
    rn_gemm_bf16_wmma<256><<<1024 / 128, 256, 0, stream>>>(f1b, wbF2, fb2, f2b, nullptr, 1024, 1);
    rn_final_fc<<<(1024 * 10 + 255) / 256, 256, 0, stream>>>(f2b, fw3, fb3, out);
}
